// Model_11330123727412
// MI455X (gfx1250) — matrix operand order repaired and checked by reading, not compiled, not run on hardware
//
#include <hip/hip_runtime.h>
#include <hip/hip_bf16.h>
#include <math.h>

// ---------------- model dims ----------------
#define BB   64
#define TT   512
#define DD   256
#define HH   256
#define CIN  32
#define COUT 32
#define PRED 48
#define NF   32
#define G4   1024   // 4*H

typedef __attribute__((ext_vector_type(16))) __bf16 v16bf;
typedef __attribute__((ext_vector_type(8)))  float  v8f;

#define DEV __device__ __forceinline__

DEV v8f wmma_bf(v16bf a, v16bf b, v8f c) {
  return __builtin_amdgcn_wmma_f32_16x16x32_bf16(false, a, false, b, (short)0, c, false, false);
}

// A fragment: 16x32 bf16, row-major source. ISA layout: lanes 0-15 row m K{0..7,16..23},
// lanes 16-31 row m K{8..15,24..31}. Two 16B loads per lane.
DEV v16bf load_a16(const __bf16* A, int lda) {
  int lane = threadIdx.x & 31;
  const __bf16* p = A + (long)(lane & 15) * lda + ((lane >> 4) << 3);
  v16bf r;
  *(uint4*)&r         = *(const uint4*)p;
  *(((uint4*)&r) + 1) = *(const uint4*)(p + 16);
  return r;
}

// B fragment from B^T stored (N,K) row-major: lane n<16 -> col n, K{0..7,16..23};
// lane n+16 -> col n, K{8..15,24..31}.
DEV v16bf load_bt16(const __bf16* BT, int ldb) {
  int lane = threadIdx.x & 31;
  const __bf16* p = BT + (long)(lane & 15) * ldb + ((lane >> 4) << 3);
  v16bf r;
  *(uint4*)&r         = *(const uint4*)p;
  *(((uint4*)&r) + 1) = *(const uint4*)(p + 16);
  return r;
}

// Async global->LDS copy, 16B per lane (GLOBAL_LOAD_ASYNC_TO_LDS_B128, ASYNCcnt path).
// Low 32 bits of a generic LDS pointer are the LDS byte offset (aperture rule, ISA 10.2).
DEV void async_b128(void* lds, const void* gl) {
  unsigned l = (unsigned)(unsigned long long)(uintptr_t)lds;
  asm volatile("global_load_async_to_lds_b128 %0, %1, off" :: "v"(l), "v"(gl) : "memory");
}
DEV void wait_async0() { asm volatile("s_wait_asynccnt 0x0" ::: "memory"); }

// ---------------- generic batched WMMA GEMM (one tile per wave) ----------------
// C(M,N) = A(M,K) @ BT(N,K)^T  (+bias).  biasMode: 0 none, 1 per-N, 2 per-M.
__global__ void __launch_bounds__(256)
k_gemm(const __bf16* __restrict__ A, long sA,
       const __bf16* __restrict__ BT, long sB,
       float* __restrict__ Cf, __bf16* __restrict__ Cb, long sC,
       const float* __restrict__ bias, long sBias, int biasMode,
       int M, int N, int K, int lda, int ldb, int ldc, int nbatch)
{
  int mt = M >> 4, nt = N >> 4;
  long tpb = (long)mt * nt;
  long tile = (long)blockIdx.x * 8 + (threadIdx.x >> 5);
  if (tile >= tpb * nbatch) return;
  int b  = (int)(tile / tpb);
  long t = tile - (long)b * tpb;
  int tm = (int)(t / nt), tn = (int)(t % nt);

  const __bf16* Ab = A  + (long)b * sA + (long)tm * 16 * lda;
  const __bf16* Bb = BT + (long)b * sB + (long)tn * 16 * ldb;

  v8f acc = {};
  for (int k = 0; k < K; k += 32) {
    if (k + 128 < K) { __builtin_prefetch(Ab + k + 128, 0, 3); __builtin_prefetch(Bb + k + 128, 0, 3); }
    v16bf av = load_a16(Ab + k, lda);
    v16bf bv = load_bt16(Bb + k, ldb);
    acc = wmma_bf(av, bv, acc);
  }
  int lane = threadIdx.x & 31, n = lane & 15, hi = lane >> 4;
  if (biasMode == 1) {
    float bv = bias[(long)b * sBias + tn * 16 + n];
    #pragma unroll
    for (int r = 0; r < 8; r++) acc[r] += bv;
  } else if (biasMode == 2) {
    #pragma unroll
    for (int r = 0; r < 8; r++) acc[r] += bias[(long)b * sBias + tm * 16 + r + hi * 8];
  }
  long cbase = (long)b * sC + (long)tm * 16 * ldc + tn * 16;
  #pragma unroll
  for (int r = 0; r < 8; r++) {
    long off = cbase + (long)(r + hi * 8) * ldc + n;
    if (Cf) Cf[off] = acc[r];
    if (Cb) Cb[off] = (__bf16)acc[r];
  }
}

// ---------------- pipelined WMMA GEMM: async-LDS B staging, 128x64 patch/block ----------
// 8 waves share a double-buffered LDS copy of B (async copy of step k+32 overlaps WMMA on
// step k).  Each wave: one A fragment reused across 4 N-tiles (4 accumulators).
// Requires N % 64 == 0.
__global__ void __launch_bounds__(256)
k_gemm_p(const __bf16* __restrict__ A, long sA,
         const __bf16* __restrict__ BT, long sB,
         float* __restrict__ Cf, __bf16* __restrict__ Cb, long sC,
         const float* __restrict__ bias, long sBias, int biasMode,
         int M, int N, int K, int lda, int ldb, int ldc, int nbatch)
{
  __shared__ __bf16 Bs[2][64 * 32];   // 2 x 4KB
  int mg = (M + 127) >> 7;            // groups of 8 m-tiles
  int ng = N >> 6;                    // groups of 4 n-tiles
  long gpb = (long)mg * ng;
  long grp = blockIdx.x;
  int b  = (int)(grp / gpb);
  long g = grp - (long)b * gpb;
  int gm = (int)(g / ng), gn = (int)(g % ng);

  int wave = threadIdx.x >> 5, lane = threadIdx.x & 31, n = lane & 15, hi = lane >> 4;
  int tm = gm * 8 + wave;
  bool act = (tm << 4) < M;           // wave-uniform

  const __bf16* Ab = A  + (long)b * sA + (long)tm * 16 * lda;
  const __bf16* Bb = BT + (long)b * sB + (long)gn * 64 * ldb;

  // producer mapping: 256 threads x 16B = 64 rows x 64B (one 64x32 bf16 B chunk)
  int prow = threadIdx.x >> 2, pseg = threadIdx.x & 3;
  const __bf16* gsrc = Bb + (long)prow * ldb + pseg * 8;
  __bf16* ldst0 = &Bs[0][prow * 32 + pseg * 8];
  __bf16* ldst1 = &Bs[1][prow * 32 + pseg * 8];

  v8f acc[4] = {{}, {}, {}, {}};
  async_b128(ldst0, gsrc);
  int cur = 0;
  for (int k = 0; k < K; k += 32) {
    wait_async0();
    __syncthreads();                       // Bs[cur] ready for the whole block
    if (k + 32 < K)
      async_b128(cur ? ldst0 : ldst1, gsrc + (k + 32));
    if (act) {
      if (k + 128 < K) __builtin_prefetch(Ab + k + 128, 0, 3);
      v16bf av = load_a16(Ab + k, lda);
      #pragma unroll
      for (int j = 0; j < 4; ++j) {
        v16bf bv = load_bt16(&Bs[cur][(j * 16) * 32], 32);
        acc[j] = wmma_bf(av, bv, acc[j]);
      }
    }
    __syncthreads();                       // done reading Bs[cur]; safe to refill next iter
    cur ^= 1;
  }
  if (!act) return;
  #pragma unroll
  for (int j = 0; j < 4; ++j) {
    int tn = gn * 4 + j;
    if (biasMode == 1) {
      float bv = bias[(long)b * sBias + tn * 16 + n];
      #pragma unroll
      for (int r = 0; r < 8; r++) acc[j][r] += bv;
    } else if (biasMode == 2) {
      #pragma unroll
      for (int r = 0; r < 8; r++) acc[j][r] += bias[(long)b * sBias + tm * 16 + r + hi * 8];
    }
    long cbase = (long)b * sC + (long)tm * 16 * ldc + tn * 16;
    #pragma unroll
    for (int r = 0; r < 8; r++) {
      long off = cbase + (long)(r + hi * 8) * ldc + n;
      if (Cf) Cf[off] = acc[j][r];
      if (Cb) Cb[off] = (__bf16)acc[j][r];
    }
  }
}

// ---------------- prep kernels ----------------
__global__ void k_cvt(const float* s, __bf16* d, long n) {
  long i = (long)blockIdx.x * 256 + threadIdx.x;
  if (i < n) d[i] = (__bf16)s[i];
}
__global__ void k_trans(const float* src, __bf16* dst, int R, int C, int dld) {
  long i = (long)blockIdx.x * 256 + threadIdx.x;
  if (i >= (long)R * C) return;
  int r = (int)(i / C), c = (int)(i % C);
  dst[(long)c * dld + r] = (__bf16)src[i];
}
// rFFT basis F(64,512): rows 0-31 cos, rows 32-63 -sin
__global__ void k_basisF(__bf16* F) {
  int k = blockIdx.x, t = threadIdx.x;
  int f = k & 31;
  float th = 6.283185307179586f * f * t / 512.f;
  F[(long)k * 512 + t] = (__bf16)((k < 32) ? cosf(th) : -sinf(th));
}
// irFFT basis G(512,64): G[t][f]=w*cos/T, G[t][32+f]=-w*sin/T
__global__ void k_basisG(__bf16* G) {
  int t = blockIdx.x, j = threadIdx.x;
  int f = j & 31;
  float w = (f == 0) ? 1.f : 2.f;
  float th = 6.283185307179586f * f * t / 512.f;
  G[(long)t * 64 + j] = (__bf16)((j < 32) ? w * cosf(th) / 512.f : -w * sinf(th) / 512.f);
}
// SFD combined weights, transposed (f, o, k): k<256 from Wr/Wi, k>=256 from -Wi/Wr
__global__ void k_wcat(const float* wr, const float* wi, __bf16* WR, __bf16* WI) {
  long i = (long)blockIdx.x * 256 + threadIdx.x;  // (f*256+o)*512+k
  if (i >= 32L * 256 * 512) return;
  int k = (int)(i & 511); long fo = i >> 9; int o = (int)(fo & 255); int f = (int)(fo >> 8);
  if (k < 256) {
    WR[i] = (__bf16)wr[((long)f * 256 + k) * 256 + o];
    WI[i] = (__bf16)wi[((long)f * 256 + k) * 256 + o];
  } else {
    int kk = k - 256;
    WR[i] = (__bf16)(-wi[((long)f * 256 + kk) * 256 + o]);
    WI[i] = (__bf16)( wr[((long)f * 256 + kk) * 256 + o]);
  }
}
// Folded decoder spectral operator MratT (o, k=16384): per-f complex matmul + irfft last row
__global__ void k_mrat(const float* wr, const float* wi, __bf16* MT) {
  long i = (long)blockIdx.x * 256 + threadIdx.x;  // o*16384 + f*512 + ri*256 + d
  if (i >= 256L * 16384) return;
  int kk = (int)(i & 16383); int o = (int)(i >> 14);
  int f = kk >> 9, rem = kk & 511, ri = rem >> 8, d = rem & 255;
  float w  = (f == 0) ? 1.f : 2.f;
  float th = 6.283185307179586f * f * 511.f / 512.f;
  float cr =  w * cosf(th) / 512.f;
  float ci = -w * sinf(th) / 512.f;
  float a = wr[((long)f * 256 + d) * 256 + o], bb = wi[((long)f * 256 + d) * 256 + o];
  MT[i] = (__bf16)((ri == 0) ? (cr * a + ci * bb) : (ci * a - cr * bb));
}
__global__ void k_cvec(const float* br, const float* bi, float* cv) {
  int o = threadIdx.x;
  float s = 0.f;
  for (int f = 0; f < 32; f++) {
    float w = (f == 0) ? 1.f : 2.f;
    float th = 6.283185307179586f * f * 511.f / 512.f;
    s += (w * cosf(th) / 512.f) * br[f * 256 + o] + (-w * sinf(th) / 512.f) * bi[f * 256 + o];
  }
  cv[o] = s;
}

// ---------------- encoder-path pack / elementwise ----------------
__global__ void k_pack_sfdA(const float* band, __bf16* Abf) {
  int f = blockIdx.x, b = blockIdx.y, k = threadIdx.x;
  float v = (k < 256) ? band[((long)b * 64 + f) * 256 + k]
                      : band[((long)b * 64 + 32 + f) * 256 + (k - 256)];
  Abf[((long)f * 64 + b) * 512 + k] = (__bf16)v;
}
__global__ void k_pack_SFT(const float* outR, const float* outI, __bf16* S) {
  int b = blockIdx.x, d = threadIdx.x;
  for (int j = 0; j < 64; j++) {
    float v = (j < 32) ? outR[((long)j * 64 + b) * 256 + d]
                       : outI[((long)(j - 32) * 64 + b) * 256 + d];
    S[((long)b * 256 + d) * 64 + j] = (__bf16)v;
  }
}
__global__ void k_pack_encA(const int* mark, const float* hol, const float* seasonT, __bf16* encA) {
  int t = blockIdx.x, b = blockIdx.y, j = threadIdx.x;
  int idx = mark[((long)b * 512 + t) * 5 + 4];
  float v = (j < 256) ? hol[(long)idx * 256 + j]
                      : seasonT[(long)b * 131072 + (long)(j - 256) * 512 + t];
  encA[((long)t * 64 + b) * 512 + j] = (__bf16)v;
}
__global__ void k_encout(const float* tmp, const float* xnetT, __bf16* eo, __bf16* prev) {
  int t = blockIdx.x, b = blockIdx.y, d = threadIdx.x;
  long r = (long)t * 64 + b;
  float v = tmp[r * 256 + d] + xnetT[(long)b * 131072 + (long)d * 512 + t];
  eo[r * 256 + d] = (__bf16)v;
  if (t == 511) prev[b * 256 + d] = (__bf16)v;
}
__global__ void k_ring_init(const float* xnetT, __bf16* ring) {
  int b = blockIdx.x, t = blockIdx.y, d = threadIdx.x;
  ring[((long)b * 512 + t) * 256 + d] = (__bf16)xnetT[(long)b * 131072 + (long)d * 512 + t];
}

// ---------------- encoder BiLSTM (persistent, 1 WG per direction) ----------------
__global__ void __launch_bounds__(1024)
k_lstm_enc(const __bf16* __restrict__ Zin, const __bf16* __restrict__ WhhT,
           __bf16* h_state, float* c_state)
{
  int dir = blockIdx.x;
  __shared__ __bf16 hbuf[64 * 256];
  for (int i = threadIdx.x; i < 64 * 256; i += 1024) hbuf[i] = (__bf16)0.f;
  __syncthreads();

  int wave = threadIdx.x >> 5, lane = threadIdx.x & 31, n = lane & 15, hi = lane >> 4;
  const __bf16* WT = WhhT + (long)dir * G4 * 256;
  v8f cacc[2]; cacc[0] = (v8f){}; cacc[1] = (v8f){};

  for (int step = 0; step < 512; ++step) {
    int t = dir ? 511 - step : step;
    v8f hnew[2];
    #pragma unroll
    for (int u2 = 0; u2 < 2; ++u2) {
      int u = wave * 2 + u2, mt = u >> 4, nt = u & 15;
      v8f acc[4] = {{}, {}, {}, {}};
      for (int k = 0; k < 256; k += 32) {
        v16bf av = load_a16(hbuf + mt * 16 * 256 + k, 256);
        #pragma unroll
        for (int g = 0; g < 4; ++g) {
          v16bf bv = load_bt16(WT + (long)(g * 256 + nt * 16) * 256 + k, 256);
          acc[g] = wmma_bf(av, bv, acc[g]);
        }
      }
      #pragma unroll
      for (int r = 0; r < 8; r++) {
        int m = mt * 16 + r + hi * 8, col = nt * 16 + n;
        const __bf16* z = Zin + ((long)dir * 32768 + (long)t * 64 + m) * G4;
        float zi = acc[0][r] + (float)z[0 * 256 + col];
        float zf = acc[1][r] + (float)z[1 * 256 + col];
        float zg = acc[2][r] + (float)z[2 * 256 + col];
        float zo = acc[3][r] + (float)z[3 * 256 + col];
        float ig = 1.f / (1.f + __expf(-zi));
        float fg = 1.f / (1.f + __expf(-zf));
        float og = 1.f / (1.f + __expf(-zo));
        float gg = tanhf(zg);
        float c  = fg * cacc[u2][r] + ig * gg;
        cacc[u2][r] = c;
        hnew[u2][r] = og * tanhf(c);
      }
    }
    __syncthreads();
    #pragma unroll
    for (int u2 = 0; u2 < 2; ++u2) {
      int u = wave * 2 + u2, mt = u >> 4, nt = u & 15;
      #pragma unroll
      for (int r = 0; r < 8; r++)
        hbuf[(mt * 16 + r + hi * 8) * 256 + nt * 16 + n] = (__bf16)hnew[u2][r];
    }
    __syncthreads();
  }
  for (int i = threadIdx.x; i < 64 * 256; i += 1024) h_state[(long)dir * 16384 + i] = hbuf[i];
  #pragma unroll
  for (int u2 = 0; u2 < 2; ++u2) {
    int u = wave * 2 + u2, mt = u >> 4, nt = u & 15;
    #pragma unroll
    for (int r = 0; r < 8; r++)
      c_state[(long)dir * 16384 + (mt * 16 + r + hi * 8) * 256 + nt * 16 + n] = cacc[u2][r];
  }
}

// ---------------- decoder step: 2 LSTM cells + Wp proj + window column proj ----------------
__global__ void __launch_bounds__(1024)
k_dec_cell(const __bf16* __restrict__ prev, __bf16* h_state, float* c_state,
           const __bf16* __restrict__ decWT, const float* __restrict__ dec_b,
           const __bf16* __restrict__ WpT, const float* __restrict__ bp,
           const float* __restrict__ Wnet, const float* __restrict__ bnet,
           float* v_buf, float* preds, int s)
{
  extern __shared__ __bf16 smem[];
  __bf16* Acat  = smem;                 // [2][64][512]
  __bf16* Hcat  = smem + 2 * 64 * 512;  // [64][512]
  float*  prevx = (float*)(Hcat + 64 * 512);  // [64][32]

  for (int i = threadIdx.x; i < 64 * 512; i += 1024) {
    int b = i >> 9, j = i & 511;
    if (j < 256) { __bf16 pv = prev[b * 256 + j]; Acat[i] = pv; Acat[32768 + i] = pv; }
    else {
      Acat[i]          = h_state[b * 256 + (j - 256)];
      Acat[32768 + i]  = h_state[16384 + b * 256 + (j - 256)];
    }
  }
  __syncthreads();
  int wave = threadIdx.x >> 5, lane = threadIdx.x & 31, n = lane & 15, hi = lane >> 4;

  for (int u = wave; u < 128; u += 32) {
    int cell = u >> 6, r6 = u & 63, mt = r6 >> 4, nt = r6 & 15;
    const __bf16* Ab = Acat + cell * 32768 + mt * 16 * 512;
    const __bf16* WT = decWT + (long)cell * G4 * 512;
    v8f acc[4] = {{}, {}, {}, {}};
    for (int k = 0; k < 512; k += 32) {
      v16bf av = load_a16(Ab + k, 512);
      #pragma unroll
      for (int g = 0; g < 4; ++g) {
        v16bf bv = load_bt16(WT + (long)(g * 256 + nt * 16) * 512 + k, 512);
        acc[g] = wmma_bf(av, bv, acc[g]);
      }
    }
    #pragma unroll
    for (int r = 0; r < 8; r++) {
      int m = mt * 16 + r + hi * 8, col = nt * 16 + n;
      float zi = acc[0][r] + dec_b[cell * G4 + 0 * 256 + col];
      float zf = acc[1][r] + dec_b[cell * G4 + 1 * 256 + col];
      float zg = acc[2][r] + dec_b[cell * G4 + 2 * 256 + col];
      float zo = acc[3][r] + dec_b[cell * G4 + 3 * 256 + col];
      float ig = 1.f / (1.f + __expf(-zi));
      float fg = 1.f / (1.f + __expf(-zf));
      float og = 1.f / (1.f + __expf(-zo));
      float gg = tanhf(zg);
      long cidx = (long)cell * 16384 + m * 256 + col;
      float c = fg * c_state[cidx] + ig * gg;
      c_state[cidx] = c;
      float h = og * tanhf(c);
      h_state[cidx] = (__bf16)h;
      Hcat[m * 512 + cell * 256 + col] = (__bf16)h;
    }
  }
  __syncthreads();
  if (wave < 8) {  // prev_x = [h0|h1] @ Wp + bp
    int mt = wave >> 1, nt = wave & 1;
    v8f acc = {};
    for (int k = 0; k < 512; k += 32) {
      v16bf av = load_a16(Hcat + mt * 16 * 512 + k, 512);
      v16bf bv = load_bt16(WpT + (long)nt * 16 * 512 + k, 512);
      acc = wmma_bf(av, bv, acc);
    }
    #pragma unroll
    for (int r = 0; r < 8; r++) {
      int m = mt * 16 + r + hi * 8, c = nt * 16 + n;
      float vv = acc[r] + bp[c];
      preds[(long)m * (PRED * COUT) + s * COUT + c] = vv;
      prevx[m * 32 + c] = vv;
    }
  }
  __syncthreads();
  // v = prev_x @ Wnet + bnet (new ring column, also pat1[:, -1])
  for (int i = threadIdx.x; i < 64 * 256; i += 1024) {
    int b = i >> 8, d = i & 255;
    float a = bnet[d];
    #pragma unroll 8
    for (int q = 0; q < 32; q++) a += prevx[b * 32 + q] * Wnet[q * 256 + d];
    v_buf[i] = a;
  }
}

// DFT shift rotation: X' = e^{i 2pi f/T} (X - old0 + v)
__global__ void k_rotate(float* band, __bf16* bandcat, const __bf16* ring, const float* v_buf, int s) {
  int b = blockIdx.x, f = blockIdx.y, d = threadIdx.x;
  float old0 = (float)ring[((long)b * 512 + (s & 511)) * 256 + d];
  float v  = v_buf[b * 256 + d];
  float xr = band[((long)b * 64 + f) * 256 + d] - old0 + v;
  float xi = band[((long)b * 64 + 32 + f) * 256 + d];
  float sn, cs; __sincosf(6.283185307179586f * f / 512.f, &sn, &cs);
  float nr = xr * cs - xi * sn, ni = xr * sn + xi * cs;
  band[((long)b * 64 + f) * 256 + d]      = nr;
  band[((long)b * 64 + 32 + f) * 256 + d] = ni;
  bandcat[(long)b * 16384 + f * 512 + d]       = (__bf16)nr;
  bandcat[(long)b * 16384 + f * 512 + 256 + d] = (__bf16)ni;
}
__global__ void k_packprev(const float* sea, const float* hol, const int* mark_dec,
                           __bf16* Aprev, __bf16* ring, const float* v_buf, int s) {
  int b = blockIdx.x, j = threadIdx.x;
  int idx = mark_dec[((long)b * PRED + s) * 5 + 4];
  float x = (j < 256) ? sea[b * 256 + j] : hol[(long)idx * 256 + (j - 256)];
  Aprev[(long)b * 512 + j] = (__bf16)x;
  if (j < 256) ring[((long)b * 512 + (s & 511)) * 256 + j] = (__bf16)v_buf[b * 256 + j];
}
__global__ void k_prevnew(const float* v_buf, const float* tmp, __bf16* prev) {
  int b = blockIdx.x, d = threadIdx.x;
  prev[b * 256 + d] = (__bf16)(v_buf[b * 256 + d] + tmp[b * 256 + d]);
}

// ---------------- contrastive branch (DC-bin shortcut) ----------------
__global__ void k_rowsum(const float* xq, const float* xk, float* rsum) {
  int b = blockIdx.x, y = blockIdx.y, c = threadIdx.x;
  const float* x = y ? xk : xq;
  float s = 0.f;
  for (int t = 0; t < 512; t++) s += x[((long)b * 512 + t) * 32 + c];
  rsum[((long)y * 64 + b) * 32 + c] = s;
}
__global__ void k_xf0(const float* rsum, const float* Wnet, const float* bnet, float* xf0) {
  int b = blockIdx.x, y = blockIdx.y, d = threadIdx.x;
  float a = 512.f * bnet[d];
  for (int i = 0; i < 32; i++) a += rsum[((long)y * 64 + b) * 32 + i] * Wnet[i * 256 + d];
  xf0[((long)y * 64 + b) * 256 + d] = a;
}
__global__ void k_qk(const float* xf0, const float* wr0, const float* br0, float* vq, float* vk) {
  int b = blockIdx.x, y = blockIdx.y, o = threadIdx.x;
  float a = br0[o];
  for (int d = 0; d < 256; d++) a += xf0[((long)y * 64 + b) * 256 + d] * wr0[d * 256 + o];
  (y ? vk : vq)[b * 256 + o] = a / 512.f;
}
__global__ void k_norm(const float* v, float* o) {
  int b = blockIdx.x, lane = threadIdx.x;
  float s = 0.f;
  for (int d = lane; d < 256; d += 32) { float x = v[b * 256 + d]; s += x * x; }
  for (int m = 16; m; m >>= 1) s += __shfl_xor(s, m, 32);
  float inv = 1.f / (sqrtf(s) + 1e-8f);
  for (int d = lane; d < 256; d += 32) o[b * 256 + d] = v[b * 256 + d] * inv;
}
__global__ void __launch_bounds__(1024) k_contra(const float* qn, const float* kn, float* out) {
  __shared__ float tot;
  if (threadIdx.x == 0) tot = 0.f;
  __syncthreads();
  int warp = threadIdx.x >> 5, lane = threadIdx.x & 31;
  float lsum = 0.f;
  for (int i = warp; i < 64; i += 32) {
    float v0 = 0.f, v1 = 0.f;
    for (int d = 0; d < 256; d++) {
      float qv = qn[i * 256 + d];
      v0 += qv * kn[lane * 256 + d];
      v1 += qv * kn[(lane + 32) * 256 + d];
    }
    v0 *= 10.f; v1 *= 10.f;  // 1/TEMP
    float mx = fmaxf(v0, v1);
    for (int m = 16; m; m >>= 1) mx = fmaxf(mx, __shfl_xor(mx, m, 32));
    float se = __expf(v0 - mx) + __expf(v1 - mx);
    for (int m = 16; m; m >>= 1) se += __shfl_xor(se, m, 32);
    float d0 = __shfl(v0, i & 31, 32), d1 = __shfl(v1, i & 31, 32);
    float diag = (i < 32) ? d0 : d1;
    if (lane == 0) lsum += diag - (mx + __logf(se));
  }
  if (lane == 0) atomicAdd(&tot, lsum);
  __syncthreads();
  if (threadIdx.x == 0) out[0] = -tot / 64.f;
}

// ---------------- host ----------------
static inline int g256(long n) { return (int)((n + 255) / 256); }

static inline void gemm(hipStream_t st, const __bf16* A, long sA, const __bf16* BT, long sB,
                        float* Cf, __bf16* Cb, long sC, const float* bias, long sBias, int bm,
                        int M, int N, int K, int lda, int ldb, int ldc, int nb) {
  long tiles = (long)(M >> 4) * (N >> 4) * nb;
  k_gemm<<<(int)((tiles + 7) / 8), 256, 0, st>>>(A, sA, BT, sB, Cf, Cb, sC, bias, sBias, bm,
                                                 M, N, K, lda, ldb, ldc, nb);
}
static inline void gemm_p(hipStream_t st, const __bf16* A, long sA, const __bf16* BT, long sB,
                          float* Cf, __bf16* Cb, long sC, const float* bias, long sBias, int bm,
                          int M, int N, int K, int lda, int ldb, int ldc, int nb) {
  long grps = (long)((M + 127) >> 7) * (N >> 6) * nb;
  k_gemm_p<<<(int)grps, 256, 0, st>>>(A, sA, BT, sB, Cf, Cb, sC, bias, sBias, bm,
                                      M, N, K, lda, ldb, ldc, nb);
}

extern "C" void kernel_launch(void* const* d_in, const int* in_sizes, int n_in,
                              void* d_out, int out_size, void* d_ws, size_t ws_size,
                              hipStream_t stream) {
  (void)in_sizes; (void)n_in; (void)out_size; (void)ws_size;
  const float* x_enc   = (const float*)d_in[0];
  const float* x_q     = (const float*)d_in[1];
  const float* x_k     = (const float*)d_in[2];
  const int*   mark_e  = (const int*)  d_in[3];
  const int*   mark_d  = (const int*)  d_in[4];
  const float* hol     = (const float*)d_in[5];
  const float* Wf      = (const float*)d_in[6];
  const float* bf      = (const float*)d_in[7];
  const float* Wnet    = (const float*)d_in[8];
  const float* bnet    = (const float*)d_in[9];
  const float* sfd_wr  = (const float*)d_in[10];
  const float* sfd_wi  = (const float*)d_in[11];
  const float* sfd_br  = (const float*)d_in[12];
  const float* sfd_bi  = (const float*)d_in[13];
  const float* eWih    = (const float*)d_in[14];
  const float* eWhh    = (const float*)d_in[15];
  const float* enc_b   = (const float*)d_in[16];
  const float* dWih    = (const float*)d_in[17];
  const float* dWhh    = (const float*)d_in[18];
  const float* dec_b   = (const float*)d_in[19];
  const float* Wp      = (const float*)d_in[20];
  const float* bp      = (const float*)d_in[21];
  float* out = (float*)d_out;

  char* w = (char*)d_ws; size_t off = 0;
  auto alloc = [&](size_t bytes) -> char* {
    char* p = w + off; off = (off + bytes + 255) & ~(size_t)255; return p;
  };
  // bf16 buffers
  __bf16* xe_bf   = (__bf16*)alloc(1048576L * 2);
  __bf16* WnetT   = (__bf16*)alloc(8192L * 2);
  __bf16* WihTe   = (__bf16*)alloc(524288L * 2);
  __bf16* WhhTe   = (__bf16*)alloc(524288L * 2);
  __bf16* decWT   = (__bf16*)alloc(1048576L * 2);
  __bf16* WfT     = (__bf16*)alloc(131072L * 2);
  __bf16* WpT     = (__bf16*)alloc(16384L * 2);
  __bf16* Fb      = (__bf16*)alloc(32768L * 2);
  __bf16* Gb      = (__bf16*)alloc(32768L * 2);
  __bf16* WcatRT  = (__bf16*)alloc(4194304L * 2);
  __bf16* WcatIT  = (__bf16*)alloc(4194304L * 2);
  __bf16* MratT   = (__bf16*)alloc(4194304L * 2);
  __bf16* xnetTbf = (__bf16*)alloc(8388608L * 2);
  __bf16* AbfSfd  = (__bf16*)alloc(1048576L * 2);
  __bf16* SFT     = (__bf16*)alloc(1048576L * 2);
  __bf16* encA    = (__bf16*)alloc(16777216L * 2);
  __bf16* eo_bf   = (__bf16*)alloc(8388608L * 2);
  __bf16* Zin     = (__bf16*)alloc(67108864L * 2);
  __bf16* ring    = (__bf16*)alloc(8388608L * 2);
  __bf16* bandcat = (__bf16*)alloc(1048576L * 2);
  __bf16* prev_bf = (__bf16*)alloc(16384L * 2);
  __bf16* Aprev   = (__bf16*)alloc(32768L * 2);
  __bf16* h_state = (__bf16*)alloc(32768L * 2);
  // f32 buffers
  float* xnetT   = (float*)alloc(8388608L * 4);
  float* band    = (float*)alloc(1048576L * 4);
  float* outR    = (float*)alloc(524288L * 4);
  float* outI    = (float*)alloc(524288L * 4);
  float* seasonT = (float*)alloc(8388608L * 4);
  float* tmp32   = (float*)alloc(8388608L * 4);
  float* c_state = (float*)alloc(32768L * 4);
  float* v_buf   = (float*)alloc(16384L * 4);
  float* seaL    = (float*)alloc(16384L * 4);
  float* cvec    = (float*)alloc(256L * 4);
  float* rsum    = (float*)alloc(4096L * 4);
  float* xf0     = (float*)alloc(32768L * 4);
  float* vq      = (float*)alloc(16384L * 4);
  float* vk      = (float*)alloc(16384L * 4);
  float* qn      = (float*)alloc(16384L * 4);
  float* kn      = (float*)alloc(16384L * 4);

  const size_t DEC_LDS = (size_t)3 * 64 * 512 * sizeof(__bf16) + 64 * 32 * sizeof(float);
  hipFuncSetAttribute((const void*)k_dec_cell, hipFuncAttributeMaxDynamicSharedMemorySize, (int)DEC_LDS);

  // ---- prep ----
  k_cvt<<<g256(1048576), 256, 0, stream>>>(x_enc, xe_bf, 1048576);
  k_trans<<<g256(32L * 256), 256, 0, stream>>>(Wnet, WnetT, 32, 256, 32);
  for (int d = 0; d < 2; d++) {
    k_trans<<<g256(256L * 1024), 256, 0, stream>>>(eWih + (long)d * 262144, WihTe + (long)d * 262144, 256, 1024, 256);
    k_trans<<<g256(256L * 1024), 256, 0, stream>>>(eWhh + (long)d * 262144, WhhTe + (long)d * 262144, 256, 1024, 256);
    k_trans<<<g256(256L * 1024), 256, 0, stream>>>(dWih + (long)d * 262144, decWT + (long)d * 524288, 256, 1024, 512);
    k_trans<<<g256(256L * 1024), 256, 0, stream>>>(dWhh + (long)d * 262144, decWT + (long)d * 524288 + 256, 256, 1024, 512);
  }
  k_trans<<<g256(512L * 256), 256, 0, stream>>>(Wf, WfT, 512, 256, 512);
  k_trans<<<g256(512L * 32), 256, 0, stream>>>(Wp, WpT, 512, 32, 512);
  k_basisF<<<64, 512, 0, stream>>>(Fb);
  k_basisG<<<512, 64, 0, stream>>>(Gb);
  k_wcat<<<g256(4194304), 256, 0, stream>>>(sfd_wr, sfd_wi, WcatRT, WcatIT);
  k_mrat<<<g256(4194304), 256, 0, stream>>>(sfd_wr, sfd_wi, MratT);
  k_cvec<<<1, 256, 0, stream>>>(sfd_br, sfd_bi, cvec);

  // ---- contrastive branch (DC-bin shortcut, no FFT needed) ----
  k_rowsum<<<dim3(64, 2), 32, 0, stream>>>(x_q, x_k, rsum);
  k_xf0<<<dim3(64, 2), 256, 0, stream>>>(rsum, Wnet, bnet, xf0);
  k_qk<<<dim3(64, 2), 256, 0, stream>>>(xf0, sfd_wr, sfd_br, vq, vk);
  k_norm<<<64, 32, 0, stream>>>(vq, qn);
  k_norm<<<64, 32, 0, stream>>>(vk, kn);
  k_contra<<<1, 1024, 0, stream>>>(qn, kn, out + (long)BB * PRED * COUT);

  // ---- encoder path ----
  // xnetT (b,256,512) = WnetT @ x_b^T
  gemm(stream, WnetT, 0, xe_bf, 16384, xnetT, xnetTbf, 131072, bnet, 0, 2,
       256, 512, 32, 32, 32, 512, 64);
  // band (b,64,256) = F @ xnet_b  (rows 0-31 Re, 32-63 Im)
  gemm(stream, Fb, 0, xnetTbf, 131072, band, nullptr, 16384, nullptr, 0, 0,
       64, 256, 512, 512, 512, 256, 64);
  k_pack_sfdA<<<dim3(32, 64), 512, 0, stream>>>(band, AbfSfd);
  gemm(stream, AbfSfd, 32768, WcatRT, 131072, outR, nullptr, 16384, sfd_br, 256, 1,
       64, 256, 512, 512, 512, 256, 32);
  gemm(stream, AbfSfd, 32768, WcatIT, 131072, outI, nullptr, 16384, sfd_bi, 256, 1,
       64, 256, 512, 512, 512, 256, 32);
  k_pack_SFT<<<64, 256, 0, stream>>>(outR, outI, SFT);
  // seasonT (b,256,512) = SFT_b @ G^T  (async-pipelined GEMM)
  gemm_p(stream, SFT, 16384, Gb, 0, seasonT, nullptr, 131072, nullptr, 0, 0,
         256, 512, 64, 64, 64, 512, 64);
  k_pack_encA<<<dim3(512, 64), 512, 0, stream>>>(mark_e, hol, seasonT, encA);
  gemm_p(stream, encA, 0, WfT, 0, tmp32, nullptr, 0, bf, 0, 1,
         32768, 256, 512, 512, 512, 256, 1);
  k_encout<<<dim3(512, 64), 256, 0, stream>>>(tmp32, xnetT, eo_bf, prev_bf);
  // Zin (2, T*B, 1024): hoisted x@Wih + bias for both directions (async-pipelined GEMM)
  gemm_p(stream, eo_bf, 0, WihTe, 262144, nullptr, Zin, 33554432, enc_b, 1024, 1,
         32768, 1024, 256, 256, 256, 1024, 2);
  k_lstm_enc<<<2, 1024, 0, stream>>>(Zin, WhhTe, h_state, c_state);
  k_ring_init<<<dim3(64, 512), 256, 0, stream>>>(xnetT, ring);

  // ---- decoder: 48 autoregressive steps ----
  for (int s = 0; s < PRED; s++) {
    k_dec_cell<<<1, 1024, DEC_LDS, stream>>>(prev_bf, h_state, c_state, decWT, dec_b,
                                             WpT, bp, Wnet, bnet, v_buf, out, s);
    k_rotate<<<dim3(64, 32), 256, 0, stream>>>(band, bandcat, ring, v_buf, s);
    gemm(stream, bandcat, 0, MratT, 0, seaL, nullptr, 0, cvec, 0, 1,
         64, 256, 16384, 16384, 16384, 256, 1);
    k_packprev<<<64, 512, 0, stream>>>(seaL, hol, mark_d, Aprev, ring, v_buf, s);
    gemm(stream, Aprev, 0, WfT, 0, tmp32, nullptr, 0, bf, 0, 1,
         64, 256, 512, 512, 512, 256, 1);
    k_prevnew<<<64, 256, 0, stream>>>(v_buf, tmp32, prev_bf);
  }
}
